// LightGCN_69733089018087
// MI455X (gfx1250) — matrix operand order repaired and checked by reading, not compiled, not run on hardware
//
#include <hip/hip_runtime.h>

#define N_USERS 100000
#define N_ITEMS 50000
#define N_NODES (N_USERS + N_ITEMS)
#define DIM 64
#define N_LAYERS 3

typedef __attribute__((ext_vector_type(16))) _Float16 v16h;
typedef __attribute__((ext_vector_type(8)))  _Float16 v8h;
typedef __attribute__((ext_vector_type(4)))  _Float16 v4h;
typedef __attribute__((ext_vector_type(8)))  float    v8f;

// ---------------------------------------------------------------------------
// Init: acc = cur = concat(user_emb, item_emb); out = 0.   (float4 granular)
// ---------------------------------------------------------------------------
__global__ void lgn_init_kernel(const float* __restrict__ ue,
                                const float* __restrict__ ie,
                                float* __restrict__ acc,
                                float* __restrict__ cur,
                                float* __restrict__ zer,
                                int total4, int user4) {
    int i = blockIdx.x * blockDim.x + threadIdx.x;
    if (i >= total4) return;
    float4 v = (i < user4) ? ((const float4*)ue)[i] : ((const float4*)ie)[i - user4];
    ((float4*)acc)[i] = v;
    ((float4*)cur)[i] = v;
    ((float4*)zer)[i] = make_float4(0.f, 0.f, 0.f, 0.f);
}

// ---------------------------------------------------------------------------
// Edge-parallel COO SpMM: 16 threads/edge, each owns one float4 dim-slice.
// Edge streams are non-temporal (keep 38.4MB embedding tables L2-resident).
// Scatter via hardware global_atomic_add_f32 (no return -> STOREcnt path).
// ---------------------------------------------------------------------------
__global__ void lgn_spmm_kernel(const int*   __restrict__ src,
                                const int*   __restrict__ dst,
                                const float* __restrict__ vals,
                                const float* __restrict__ xin,
                                float*       __restrict__ yout,
                                int n_edges) {
    long long t = (long long)blockIdx.x * blockDim.x + threadIdx.x;
    int e = (int)(t >> 4);
    if (e >= n_edges) return;
    int s4 = ((int)t & 15) * 4;

    int   se = __builtin_nontemporal_load(src  + e);
    int   de = __builtin_nontemporal_load(dst  + e);
    float w  = __builtin_nontemporal_load(vals + e);

    float4 x = *(const float4*)(xin + (long long)se * DIM + s4);
    float* y = yout + (long long)de * DIM + s4;

    unsafeAtomicAdd(y + 0, w * x.x);
    unsafeAtomicAdd(y + 1, w * x.y);
    unsafeAtomicAdd(y + 2, w * x.z);
    unsafeAtomicAdd(y + 3, w * x.w);
}

// ---------------------------------------------------------------------------
// acc += cur; and zero the buffer that becomes the next layer's output.
// ---------------------------------------------------------------------------
__global__ void lgn_accum_zero_kernel(float* __restrict__ acc,
                                      const float* __restrict__ cur,
                                      float* __restrict__ nxt,
                                      int total4) {
    int i = blockIdx.x * blockDim.x + threadIdx.x;
    if (i >= total4) return;
    float4 a = ((float4*)acc)[i];
    float4 c = ((const float4*)cur)[i];
    a.x += c.x; a.y += c.y; a.z += c.z; a.w += c.w;
    ((float4*)acc)[i] = a;
    ((float4*)nxt)[i] = make_float4(0.f, 0.f, 0.f, 0.f);
}

// ---------------------------------------------------------------------------
// Gather the 8192 user rows / item rows from acc, convert f32 -> f16, pack
// row-major [B][64] for the WMMA dot kernel. 16 threads per pair.
// ---------------------------------------------------------------------------
__global__ void lgn_gather_pack_kernel(const float* __restrict__ acc,
                                       const int* __restrict__ users,
                                       const int* __restrict__ items,
                                       _Float16* __restrict__ Uh,
                                       _Float16* __restrict__ Vh,
                                       int B) {
    int t = blockIdx.x * blockDim.x + threadIdx.x;
    int b = t >> 4;
    if (b >= B) return;
    int d4 = (t & 15) * 4;

    int u  = users[b];
    int it = items[b];

    float4 uv = *(const float4*)(acc + (long long)u * DIM + d4);
    float4 iv = *(const float4*)(acc + ((long long)N_USERS + it) * DIM + d4);

    v4h hu = { (_Float16)uv.x, (_Float16)uv.y, (_Float16)uv.z, (_Float16)uv.w };
    v4h hv = { (_Float16)iv.x, (_Float16)iv.y, (_Float16)iv.z, (_Float16)iv.w };
    *(v4h*)(Uh + b * DIM + d4) = hu;
    *(v4h*)(Vh + b * DIM + d4) = hv;
}

// ---------------------------------------------------------------------------
// Batched dot via v_wmma_f32_16x16x32_f16.
// One wave handles 16 pairs: D = U_tile (16x64) x V_tile^T (64x16) as two
// K=32 WMMA steps; gamma = diag(D) * (1/16)  [folds the two /4 scalings].
//
// CDNA5 16-bit A layout (16x32): lane l (m = l&15):
//   VGPR0-3 : K = (l<16 ? 0..7  : 8..15)   + k0
//   VGPR4-7 : K = (l<16 ? 16..23: 24..31)  + k0
// B (32x16) as loaded here: lane l holds column n = l&15, K = (l<16 ? 0..15 : 16..31)+k0; A rows use the same K order
// C/D layout: VGPR r -> M=r (lanes 0-15, N=lane) / M=r+8 (lanes 16-31, N=lane-16)
// ---------------------------------------------------------------------------
__device__ __forceinline__ float lgn_sel8(v8f c, int i) {
    float r = c[0];
#pragma unroll
    for (int k = 1; k < 8; ++k) r = (i == k) ? c[k] : r;
    return r;
}

__global__ void lgn_wmma_dot_kernel(const _Float16* __restrict__ Uh,
                                    const _Float16* __restrict__ Vh,
                                    float* __restrict__ out, int B) {
    int wave = (int)((blockIdx.x * (long long)blockDim.x + threadIdx.x) >> 5);
    int lane = threadIdx.x & 31;
    int p0 = wave * 16;
    if (p0 >= B) return;                 // wave-uniform: EXEC stays all-1s

    int m  = lane & 15;
    int hi = lane >> 4;                  // which lane-half

    const _Float16* urow = Uh + (long long)(p0 + m) * DIM;
    const _Float16* vrow = Vh + (long long)(p0 + m) * DIM;

    v8f c = {};
#pragma unroll
    for (int kb = 0; kb < 2; ++kb) {
        int k0 = kb * 32;
        v8h a_lo = *(const v8h*)(urow + k0 + hi * 16);       // K 0..7 / 16..23
        v8h a_hi = *(const v8h*)(urow + k0 + hi * 16 + 8);   // K 8..15 / 24..31
        v16h bv  = *(const v16h*)(vrow + k0 + hi * 16);      // K 0..15 / 16..31
        v16h a;
#pragma unroll
        for (int i = 0; i < 8; ++i) { a[i] = a_lo[i]; a[i + 8] = a_hi[i]; }
        c = __builtin_amdgcn_wmma_f32_16x16x32_f16(
                /*neg_a=*/false, a, /*neg_b=*/false, bv,
                /*c_mod=*/(short)0, c, /*reuse_a=*/false, /*reuse_b=*/false);
    }

    // Diagonal extraction: M==N==d.  d<8  -> lane d,    VGPR d
    //                                d>=8 -> lane 16+d-8+8=24.., VGPR d-8
    const float scale = 0.0625f;         // (1/4)*(1/4) LightGCN mean scaling
    if (lane < 8) {
        out[p0 + lane] = lgn_sel8(c, lane) * scale;
    } else if (lane >= 24) {
        out[p0 + (lane - 16)] = lgn_sel8(c, lane - 24) * scale;
    }
}

// ---------------------------------------------------------------------------
// Host-side orchestration
// ---------------------------------------------------------------------------
extern "C" void kernel_launch(void* const* d_in, const int* in_sizes, int n_in,
                              void* d_out, int out_size, void* d_ws, size_t ws_size,
                              hipStream_t stream) {
    const int*   users    = (const int*)  d_in[0];
    const int*   items    = (const int*)  d_in[1];
    const int*   edge_src = (const int*)  d_in[2];
    const int*   edge_dst = (const int*)  d_in[3];
    const float* edge_val = (const float*)d_in[4];
    const float* user_emb = (const float*)d_in[5];
    const float* item_emb = (const float*)d_in[6];
    float*       gamma    = (float*)      d_out;

    const int B       = in_sizes[0];     // 8192
    const int n_edges = in_sizes[2];     // 10M

    const long long NELEM = (long long)N_NODES * DIM;   // 9.6M floats
    float* acc  = (float*)d_ws;
    float* buf0 = acc  + NELEM;
    float* buf1 = buf0 + NELEM;
    _Float16* Uh = (_Float16*)(buf1 + NELEM);
    _Float16* Vh = Uh + (long long)B * DIM;

    const int total4 = (int)(NELEM / 4);                 // 2.4M float4
    const int user4  = (N_USERS * DIM) / 4;

    // acc = cur(buf0) = embeddings; buf1 = 0
    lgn_init_kernel<<<(total4 + 255) / 256, 256, 0, stream>>>(
        user_emb, item_emb, acc, buf0, buf1, total4, user4);

    float* cin  = buf0;
    float* cout = buf1;
    const long long spmm_threads = (long long)n_edges * 16;
    const int spmm_blocks = (int)((spmm_threads + 255) / 256);

    for (int l = 0; l < N_LAYERS; ++l) {
        lgn_spmm_kernel<<<spmm_blocks, 256, 0, stream>>>(
            edge_src, edge_dst, edge_val, cin, cout, n_edges);
        // acc += cout; zero cin (it becomes next layer's output buffer)
        lgn_accum_zero_kernel<<<(total4 + 255) / 256, 256, 0, stream>>>(
            acc, cout, cin, total4);
        float* tmp = cin; cin = cout; cout = tmp;
    }

    lgn_gather_pack_kernel<<<(B * 16 + 255) / 256, 256, 0, stream>>>(
        acc, users, items, Uh, Vh, B);

    const int waves = B / 16;                            // 512 waves
    lgn_wmma_dot_kernel<<<(waves * 32 + 255) / 256, 256, 0, stream>>>(
        Uh, Vh, gamma, B);
}
